// RegionProposalNetwork_60670708023374
// MI455X (gfx1250) — compile-verified, matrix operand order checked by reading, not run on hardware
//
#include <hip/hip_runtime.h>
#include <hip/hip_bf16.h>

// ---------------------------------------------------------------------------
// RPN pipeline for MI455X (gfx1250, wave32).
//   conv3x3 (256->256, 48x48) -> bf16 WMMA implicit GEMM, operands staged in
//     LDS by the Tensor Data Mover (tensor_load_to_lds, TENSORcnt), double-
//     buffered so TDM DMA of chunk c+1 overlaps ds_load+WMMA of chunk c.
//   1x1 heads + box decode    -> VALU kernel
//   sort by score             -> single-WG bitonic sort in LDS
//   greedy NMS                -> IoU bitmask + single-wave32 serial resolve
// ---------------------------------------------------------------------------

#define HH      48
#define WW      48
#define CIN     256
#define COUT    256
#define NPOS    (HH * WW)          // 2304
#define KDIM    (CIN * 9)          // 2304
#define MBOX    (3 * NPOS)         // 6912
#define NWORDS  216                // ceil(MBOX/32)
#define SORTN   8192
#define TOPK    2000
#define NEGV    (-1000000000.0f)
#define IMGSZ   768.0f
#define OBJTHR  0.3f
#define IOUTHR  0.7f

// GEMM tiling: block = 8 waves = 128(M) x 64(N); K chunks of 64.
#define KC       64
#define NCHUNK   (KDIM / KC)        // 36
#define A_ROWS   128
#define B_ROWS   64
#define ROWB     144                // LDS row stride: 128B data + 16B TDM pad
#define A_BUF_SZ (A_ROWS * ROWB)    // 18432
#define B_BUF_SZ (B_ROWS * ROWB)    // 9216
#define SMEM_SZ  (2 * A_BUF_SZ + 2 * B_BUF_SZ)   // 55296 (<320KB WGP LDS)

typedef float        v8f   __attribute__((ext_vector_type(8)));
typedef __bf16       v16bf __attribute__((ext_vector_type(16)));
typedef __bf16       v8bf  __attribute__((ext_vector_type(8)));
typedef unsigned int u32x4 __attribute__((ext_vector_type(4)));
typedef int          i32x8 __attribute__((ext_vector_type(8)));
typedef int          i32x4 __attribute__((ext_vector_type(4)));

#if __has_builtin(__builtin_amdgcn_tensor_load_to_lds) && \
    __has_builtin(__builtin_amdgcn_s_wait_tensorcnt)
#define USE_TDM 1
#else
#define USE_TDM 0
#endif

static __device__ __forceinline__ v16bf cat8(v8bf lo, v8bf hi) {
  return __builtin_shufflevector(lo, hi, 0,1,2,3,4,5,6,7,8,9,10,11,12,13,14,15);
}

// float -> bf16 bits, round-to-nearest-even (pure integer path)
static __device__ __forceinline__ unsigned short f2bf(float f) {
  unsigned u = __float_as_uint(f);
  u += 0x7FFFu + ((u >> 16) & 1u);
  return (unsigned short)(u >> 16);
}

// descending-order sortable key: ascending u32 order == descending float order
static __device__ __forceinline__ unsigned sortkey_desc(float f) {
  unsigned b = __float_as_uint(f);
  unsigned asc = b ^ ((b >> 31) ? 0xFFFFFFFFu : 0x80000000u);
  return ~asc;
}

#if USE_TDM
// Issue one TDM 2-D tile load: tile_dim0 = 64 bf16 (128B, K-contiguous),
// tile_dim1 = tile_rows, row stride = KDIM elements, LDS padded 16B per 128B
// row (-> 144B LDS row stride, conflict-free ds_load_b128 fragment reads).
// D# bitfields per CDNA5 ISA.
static __device__ __forceinline__ void tdm_issue(unsigned lds_addr,
                                                 const void* gptr,
                                                 unsigned tile_rows,
                                                 unsigned tensor_rows) {
  unsigned long long ga = (unsigned long long)gptr;
  u32x4 g0;
  g0.x = 1u;                                         // count = 1 (valid D#)
  g0.y = lds_addr;                                   // LDS byte address
  g0.z = (unsigned)(ga & 0xFFFFFFFFu);               // global_addr[31:0]
  g0.w = (unsigned)((ga >> 32) & 0x1FFFFFFu)         // global_addr[56:32]
         | (2u << 30);                               // type = 2 ("image")
  i32x8 g1;
  g1[0] = (int)((1u << 16)        // data_size = 1 -> 2 bytes
              | (1u << 20)        // pad_enable
              | (4u << 22)        // pad_interval: after 8*2^4 = 128 bytes
              | (3u << 25));      // pad_amount: 4 DWORDs = 16 bytes
  g1[1] = (int)((unsigned)KDIM << 16);               // tensor_dim0[15:0]
  g1[2] = (int)((tensor_rows & 0xFFFFu) << 16);      // tensor_dim1[15:0]
  g1[3] = (int)(64u << 16);                          // tile_dim0 = 64 elems
  g1[4] = (int)(tile_rows & 0xFFFFu);                // tile_dim1
  g1[5] = (int)KDIM;                                 // tensor_dim0_stride
  g1[6] = 0;
  g1[7] = 0;
  i32x4 z4 = {0, 0, 0, 0};                           // groups 2/3 unused (2-D)
#if __clang_major__ >= 23
  i32x8 z8 = {0, 0, 0, 0, 0, 0, 0, 0};
  __builtin_amdgcn_tensor_load_to_lds(g0, g1, z4, z4, z8, 0);
#else
  __builtin_amdgcn_tensor_load_to_lds(g0, g1, z4, z4, 0);
#endif
}
#endif

// --------------------------- prep: W -> bf16 -------------------------------
__global__ __launch_bounds__(256) void k_wcast(const float* __restrict__ w,
                                               unsigned short* __restrict__ wb) {
  int t = blockIdx.x * 256 + threadIdx.x;
  if (t < COUT * KDIM) wb[t] = f2bf(w[t]);
}

// --------------------------- prep: im2col^T --------------------------------
// XcT[p][k], p = h*48+w, k = c*9+kh*3+kw (K-contiguous per position row)
__global__ __launch_bounds__(256) void k_im2col(const float* __restrict__ x,
                                                unsigned short* __restrict__ xc) {
  int t = blockIdx.x * 256 + threadIdx.x;      // t = p*256 + c
  if (t >= NPOS * CIN) return;
  int p = t >> 8, c = t & 255;
  int h = p / WW, w = p % WW;
  unsigned short* dst = xc + (size_t)p * KDIM + c * 9;
  const float* src = x + (size_t)c * NPOS;
#pragma unroll
  for (int kh = 0; kh < 3; ++kh) {
    int hh = h + kh - 1;
#pragma unroll
    for (int kw = 0; kw < 3; ++kw) {
      int ww2 = w + kw - 1;
      float v = (hh >= 0 && hh < HH && ww2 >= 0 && ww2 < WW) ? src[hh * WW + ww2] : 0.0f;
      dst[kh * 3 + kw] = f2bf(v);
    }
  }
}

// --------------------------- WMMA GEMM (TDM -> LDS -> WMMA) ----------------
// D[m,n] = sum_k A[m,k]*B[n,k] + bias[m].  A = Wb (256x2304 bf16),
// B = XcT (2304x2304 bf16), both K-contiguous.  Per 64-K chunk the TDM DMAs
// a 128x64 A tile and a 64x64 B tile into LDS (double-buffered, padded rows);
// each wave then builds the documented 16-bit 16x32 fragments with
// ds_load_b128 and runs 8 WMMAs (4 N-subtiles x 2 K-steps).
__global__ __launch_bounds__(256) void k_gemm(const unsigned short* __restrict__ a_,
                                              const unsigned short* __restrict__ b_,
                                              const float* __restrict__ bias,
                                              float* __restrict__ cc) {
  __shared__ __attribute__((aligned(16))) char smem[SMEM_SZ];
  const int tid   = threadIdx.x;
  const int lane  = tid & 31;
  const int wv    = tid >> 5;                 // wave 0..7 -> M subtile
  const int mhalf = blockIdx.x / 36;          // 0..1  (M half of 128)
  const int slab  = blockIdx.x % 36;          // N slab of 64
  const int mr    = lane & 15;
  const int koffb = (lane >> 4) * 16;         // K phase, bytes

  const char* Ag = (const char*)a_;
  const char* Bg = (const char*)b_;
#if USE_TDM
  const unsigned smem_lds = (unsigned)(unsigned long long)&smem[0];
#endif

  // chunk loader: TDM path (wave 0 issues 2 DMAs) or cooperative fallback
  auto load_chunk = [&](int buf, int c) {
#if USE_TDM
    if (wv == 0) {
      tdm_issue(smem_lds + buf * A_BUF_SZ,
                Ag + ((size_t)(mhalf * 128) * KDIM + c * KC) * 2, A_ROWS, COUT);
      tdm_issue(smem_lds + 2 * A_BUF_SZ + buf * B_BUF_SZ,
                Bg + ((size_t)(slab * 64) * KDIM + c * KC) * 2, B_ROWS, NPOS);
    }
#else
    for (int idx = tid; idx < (A_ROWS + B_ROWS) * 8; idx += 256) {
      const char* s;
      char* d;
      if (idx < A_ROWS * 8) {
        int row = idx >> 3, seg = idx & 7;
        s = Ag + ((size_t)(mhalf * 128 + row) * KDIM + c * KC) * 2 + seg * 16;
        d = smem + buf * A_BUF_SZ + row * ROWB + seg * 16;
      } else {
        int j = idx - A_ROWS * 8;
        int row = j >> 3, seg = j & 7;
        s = Bg + ((size_t)(slab * 64 + row) * KDIM + c * KC) * 2 + seg * 16;
        d = smem + 2 * A_BUF_SZ + buf * B_BUF_SZ + row * ROWB + seg * 16;
      }
      *(float4*)d = *(const float4*)s;
    }
#endif
  };

  v8f acc0 = {}, acc1 = {}, acc2 = {}, acc3 = {};
  load_chunk(0, 0);

  for (int c = 0; c < NCHUNK; ++c) {
    const int cur = c & 1;
    if (c + 1 < NCHUNK) {
      load_chunk(cur ^ 1, c + 1);
#if USE_TDM
      if (wv == 0) __builtin_amdgcn_s_wait_tensorcnt(2);  // cur's 2 DMAs done
#endif
    } else {
#if USE_TDM
      if (wv == 0) __builtin_amdgcn_s_wait_tensorcnt(0);
#endif
    }
    __syncthreads();                       // publish cur buffer to all waves

    const char* Ab = smem + cur * A_BUF_SZ + (wv * 16 + mr) * ROWB + koffb;
    const char* Bb = smem + 2 * A_BUF_SZ + cur * B_BUF_SZ + mr * ROWB + koffb;
#pragma unroll
    for (int half = 0; half < 2; ++half) {
      const int kb = half * 64;            // 32 elements * 2B per K-step
      v16bf a  = cat8(*(const v8bf*)(Ab + kb), *(const v8bf*)(Ab + kb + 32));
      v16bf b0 = cat8(*(const v8bf*)(Bb + 0 * 16 * ROWB + kb),
                      *(const v8bf*)(Bb + 0 * 16 * ROWB + kb + 32));
      v16bf b1 = cat8(*(const v8bf*)(Bb + 1 * 16 * ROWB + kb),
                      *(const v8bf*)(Bb + 1 * 16 * ROWB + kb + 32));
      v16bf b2 = cat8(*(const v8bf*)(Bb + 2 * 16 * ROWB + kb),
                      *(const v8bf*)(Bb + 2 * 16 * ROWB + kb + 32));
      v16bf b3 = cat8(*(const v8bf*)(Bb + 3 * 16 * ROWB + kb),
                      *(const v8bf*)(Bb + 3 * 16 * ROWB + kb + 32));
      acc0 = __builtin_amdgcn_wmma_f32_16x16x32_bf16(false, a, false, b0, (short)0, acc0, false, false);
      acc1 = __builtin_amdgcn_wmma_f32_16x16x32_bf16(false, a, false, b1, (short)0, acc1, false, false);
      acc2 = __builtin_amdgcn_wmma_f32_16x16x32_bf16(false, a, false, b2, (short)0, acc2, false, false);
      acc3 = __builtin_amdgcn_wmma_f32_16x16x32_bf16(false, a, false, b3, (short)0, acc3, false, false);
    }
    __syncthreads();                       // all reads of cur done before reuse
  }

  // C/D layout: VGPR r -> M = r + (lane>=16 ? 8 : 0), N = lane&15
  const int mb = mhalf * 128 + wv * 16 + (lane >> 4) * 8;
  const int n0 = slab * 64 + (lane & 15);
#pragma unroll
  for (int r = 0; r < 8; ++r) {
    float bv = bias[mb + r];
    float* crow = cc + (size_t)(mb + r) * NPOS + n0;
    crow[0]  = acc0[r] + bv;
    crow[16] = acc1[r] + bv;
    crow[32] = acc2[r] + bv;
    crow[48] = acc3[r] + bv;
  }
}

// --------------------------- heads + box decode ----------------------------
__global__ __launch_bounds__(256) void k_heads(const float* __restrict__ inter,
                                               const float* __restrict__ dw,
                                               const float* __restrict__ db,
                                               const float* __restrict__ rw,
                                               const float* __restrict__ rb,
                                               const float* __restrict__ anc,
                                               float* __restrict__ boxes,
                                               float* __restrict__ sArr,
                                               unsigned* __restrict__ vArr) {
  int p = blockIdx.x * 256 + threadIdx.x;
  if (p >= NPOS) return;
  float o[3], r[12];
#pragma unroll
  for (int j = 0; j < 3; ++j) o[j] = db[j];
#pragma unroll
  for (int j = 0; j < 12; ++j) r[j] = rb[j];
  for (int cch = 0; cch < CIN; ++cch) {
    float v = inter[(size_t)cch * NPOS + p];
#pragma unroll
    for (int j = 0; j < 3; ++j) o[j] = fmaf(v, dw[j * CIN + cch], o[j]);
#pragma unroll
    for (int j = 0; j < 12; ++j) r[j] = fmaf(v, rw[j * CIN + cch], r[j]);
  }
#pragma unroll
  for (int a = 0; a < 3; ++a) {
    int i = a * NPOS + p;
    float score = 1.0f / (1.0f + expf(-o[a]));
    float dx = r[a], dy = r[3 + a], dwv = r[6 + a], dhv = r[9 + a];
    float x1 = anc[4 * i + 0], y1 = anc[4 * i + 1];
    float x2 = anc[4 * i + 2], y2 = anc[4 * i + 3];
    float aw = x2 - x1, ah = y2 - y1;
    float acx = y1 + aw * 0.5f;   // reference swaps x/y here; replicate exactly
    float acy = x1 + ah * 0.5f;
    float px = acx + dx * aw, py = acy + dy * ah;
    float pw = aw * expf(dwv), ph = ah * expf(dhv);
    float b0 = fminf(fmaxf(px - pw * 0.5f, 0.0f), IMGSZ);
    float b1 = fminf(fmaxf(py - ph * 0.5f, 0.0f), IMGSZ);
    float b2 = fminf(fmaxf(px + pw * 0.5f, 0.0f), IMGSZ);
    float b3 = fminf(fmaxf(py + ph * 0.5f, 0.0f), IMGSZ);
    bool valid = (b2 - b0 > 0.0f) && (b3 - b1 > 0.0f) && (score > OBJTHR);
    boxes[4 * i + 0] = b0; boxes[4 * i + 1] = b1;
    boxes[4 * i + 2] = b2; boxes[4 * i + 3] = b3;
    sArr[i] = valid ? score : NEGV;
    vArr[i] = valid ? 1u : 0u;
  }
}

// --------------------------- bitonic sort (1 WG, 48KB LDS) -----------------
__global__ __launch_bounds__(1024) void k_sort(const float* __restrict__ s,
                                               const unsigned* __restrict__ valid,
                                               const float* __restrict__ boxes,
                                               float* __restrict__ ssort,
                                               unsigned* __restrict__ vsort,
                                               float* __restrict__ bsort) {
  __shared__ unsigned skey[SORTN];
  __shared__ unsigned short sidx[SORTN];
  int tid = threadIdx.x;
  for (int i = tid; i < SORTN; i += 1024) {
    if (i < MBOX) { skey[i] = sortkey_desc(s[i]); sidx[i] = (unsigned short)i; }
    else          { skey[i] = 0xFFFFFFFFu;        sidx[i] = 0; }
  }
  __syncthreads();
  for (unsigned k = 2; k <= SORTN; k <<= 1) {
    for (unsigned j = k >> 1; j > 0; j >>= 1) {
      for (unsigned i = tid; i < SORTN; i += 1024) {
        unsigned ixj = i ^ j;
        if (ixj > i) {
          bool up = ((i & k) == 0);
          unsigned ki = skey[i], kj = skey[ixj];
          if ((ki > kj) == up) {
            skey[i] = kj; skey[ixj] = ki;
            unsigned short t = sidx[i]; sidx[i] = sidx[ixj]; sidx[ixj] = t;
          }
        }
      }
      __syncthreads();
    }
  }
  for (int i = tid; i < MBOX; i += 1024) {
    int o = sidx[i];
    ssort[i] = s[o];
    vsort[i] = valid[o];
    ((float4*)bsort)[i] = ((const float4*)boxes)[o];
  }
}

// --------------------------- IoU suppression mask --------------------------
__global__ __launch_bounds__(256) void k_mask(const float* __restrict__ bsort,
                                              unsigned* __restrict__ mask) {
  int t = blockIdx.x * 256 + threadIdx.x;
  if (t >= MBOX * NWORDS) return;
  int row = t / NWORDS, wb = t % NWORDS;
  float4 bi = ((const float4*)bsort)[row];
  float ai = (bi.z - bi.x) * (bi.w - bi.y);
  unsigned bits = 0;
  int j0 = wb * 32;
#pragma unroll 4
  for (int tt = 0; tt < 32; ++tt) {
    int j = j0 + tt;
    if (j > row && j < MBOX) {
      float4 bj = ((const float4*)bsort)[j];
      float aj = (bj.z - bj.x) * (bj.w - bj.y);
      float iw = fmaxf(fminf(bi.z, bj.z) - fmaxf(bi.x, bj.x), 0.0f);
      float ih = fmaxf(fminf(bi.w, bj.w) - fmaxf(bi.y, bj.y), 0.0f);
      float inter = iw * ih;
      float iou = inter / fmaxf(ai + aj - inter, 1e-9f);
      if (iou > IOUTHR) bits |= (1u << tt);
    }
  }
  mask[t] = bits;
}

// --------------------------- output fill -----------------------------------
__global__ __launch_bounds__(256) void k_initout(float* __restrict__ out) {
  int i = blockIdx.x * 256 + threadIdx.x;
  if (i < TOPK + 4 * TOPK) out[i] = (i < TOPK) ? -1.0f : 0.0f;
}

// --------------------------- serial NMS resolve (one wave32) ---------------
__global__ __launch_bounds__(32) void k_nms(const unsigned* __restrict__ mask,
                                            const unsigned* __restrict__ vsort,
                                            const float* __restrict__ ssort,
                                            const float* __restrict__ bsort,
                                            float* __restrict__ out) {
  int lane = threadIdx.x;
  unsigned rem[7];
#pragma unroll
  for (int s2 = 0; s2 < 7; ++s2) rem[s2] = 0u;
  int cnt = 0;
  for (int i = 0; i < MBOX; ++i) {
    int w = i >> 5;
    unsigned word = (lane == (w & 31)) ? rem[w >> 5] : 0u;
    word = (unsigned)__shfl((int)word, w & 31, 32);
    bool keep = (vsort[i] != 0u) && !((word >> (i & 31)) & 1u);
    if (keep) {
      const unsigned* mrow = mask + (size_t)i * NWORDS;
#pragma unroll
      for (int s2 = 0; s2 < 7; ++s2) {
        int w2 = s2 * 32 + lane;
        if (w2 < NWORDS) rem[s2] |= mrow[w2];
      }
      if (lane == 0 && cnt < TOPK) {
        out[cnt] = ssort[i];
        ((float4*)(out + TOPK))[cnt] = ((const float4*)bsort)[i];
      }
      ++cnt;
    }
  }
}

// --------------------------- host launcher ---------------------------------
extern "C" void kernel_launch(void* const* d_in, const int* in_sizes, int n_in,
                              void* d_out, int out_size, void* d_ws, size_t ws_size,
                              hipStream_t stream) {
  const float* x   = (const float*)d_in[0];
  const float* cw  = (const float*)d_in[1];
  const float* cb  = (const float*)d_in[2];
  const float* dw  = (const float*)d_in[3];
  const float* db  = (const float*)d_in[4];
  const float* rw  = (const float*)d_in[5];
  const float* rb  = (const float*)d_in[6];
  const float* anc = (const float*)d_in[7];

  char* ws = (char*)d_ws;
  // workspace layout (all 256B-aligned), total ~19.6 MB
  unsigned short* Wb    = (unsigned short*)(ws + 0);          // 256*2304*2
  unsigned short* Xc    = (unsigned short*)(ws + 1179648);    // 2304*2304*2
  float*          inter = (float*)        (ws + 11796480);    // 256*2304*4
  float*          boxes = (float*)        (ws + 14155776);    // 6912*4*4
  float*          sArr  = (float*)        (ws + 14266368);    // 6912*4
  unsigned*       vArr  = (unsigned*)     (ws + 14294016);    // 6912*4
  float*          ssort = (float*)        (ws + 14321664);    // 6912*4
  unsigned*       vsort = (unsigned*)     (ws + 14349312);    // 6912*4
  float*          bsort = (float*)        (ws + 14376960);    // 6912*4*4
  unsigned*       mask  = (unsigned*)     (ws + 14487552);    // 6912*216*4

  float* out = (float*)d_out;

  k_wcast  <<<(COUT * KDIM + 255) / 256, 256, 0, stream>>>(cw, Wb);
  k_im2col <<<(NPOS * CIN + 255) / 256, 256, 0, stream>>>(x, Xc);
  k_gemm   <<<72, 256, 0, stream>>>(Wb, Xc, cb, inter);
  k_heads  <<<(NPOS + 255) / 256, 256, 0, stream>>>(inter, dw, db, rw, rb, anc,
                                                    boxes, sArr, vArr);
  k_sort   <<<1, 1024, 0, stream>>>(sArr, vArr, boxes, ssort, vsort, bsort);
  k_mask   <<<(MBOX * NWORDS + 255) / 256, 256, 0, stream>>>(bsort, mask);
  k_initout<<<(5 * TOPK + 255) / 256, 256, 0, stream>>>(out);
  k_nms    <<<1, 32, 0, stream>>>(mask, vsort, ssort, bsort, out);
}
